// OPTCacheFlowAttention_20272245637424
// MI455X (gfx1250) — compile-verified, matrix operand order checked by reading, not run on hardware
//
#include <hip/hip_runtime.h>
#include <hip/hip_bf16.h>

typedef __attribute__((ext_vector_type(16))) _Float16 v16h;
typedef __attribute__((ext_vector_type(8)))  float    v8f;

#define SCALE_F 0.08838834764831845f
#define NUM_PROMPTS 2
#define PROMPT_LEN 2048
#define NUM_DECODE 32
#define NUM_HEADS 16
#define HEAD_SIZE 128
#define NUM_BLOCKS 2048
#define BLOCK_SZ 16
#define MAX_BPS 64
#define HD (NUM_HEADS * HEAD_SIZE)            /* 2048 */
#define P_TOKENS (NUM_PROMPTS * PROMPT_LEN)   /* 4096 */
#define T_TOKENS (P_TOKENS + NUM_DECODE)      /* 4128 */
#define NEG_BIG (-3.0e38f)

#define WAVES_PB 2     /* waves per prefill block */
#define VROW 132       /* padded floats per key row in LDS (16B aligned, bank-safe) */

#if __has_builtin(__builtin_amdgcn_global_load_async_to_lds_b128) && \
    __has_builtin(__builtin_amdgcn_s_wait_asynccnt)
#define USE_ASYNC 1
#else
#define USE_ASYNC 0
#endif

// builtin takes pointers to 4-int vectors: (int4 AS1*, int4 AS3*, imm, imm)
typedef int vi4 __attribute__((vector_size(4 * sizeof(int))));
typedef __attribute__((address_space(1))) vi4 gvi4;
typedef __attribute__((address_space(3))) vi4 lvi4;

__device__ __forceinline__ v8f zero8() {
  v8f z;
#pragma unroll
  for (int i = 0; i < 8; ++i) z[i] = 0.0f;
  return z;
}

// Load a 16x32 (rows x k-dims) 16-bit fragment slice for this lane.
// base points at THIS LANE's row: &x[tok*HD + h*HEAD_SIZE + 32*c].
// hi = lane>=16. Halves 0..7 <- dims (8*hi + 0..7); halves 8..15 <- dims (16 + 8*hi + 0..7).
__device__ __forceinline__ v16h load_frag(const float* __restrict__ base, int hi, float scale) {
  const float4* p0 = reinterpret_cast<const float4*>(base + 8 * hi);
  const float4* p1 = reinterpret_cast<const float4*>(base + 16 + 8 * hi);
  float4 a0 = p0[0], a1 = p0[1];
  float4 b0 = p1[0], b1 = p1[1];
  v16h f;
  f[0]  = (_Float16)(a0.x * scale); f[1]  = (_Float16)(a0.y * scale);
  f[2]  = (_Float16)(a0.z * scale); f[3]  = (_Float16)(a0.w * scale);
  f[4]  = (_Float16)(a1.x * scale); f[5]  = (_Float16)(a1.y * scale);
  f[6]  = (_Float16)(a1.z * scale); f[7]  = (_Float16)(a1.w * scale);
  f[8]  = (_Float16)(b0.x * scale); f[9]  = (_Float16)(b0.y * scale);
  f[10] = (_Float16)(b0.z * scale); f[11] = (_Float16)(b0.w * scale);
  f[12] = (_Float16)(b1.x * scale); f[13] = (_Float16)(b1.y * scale);
  f[14] = (_Float16)(b1.z * scale); f[15] = (_Float16)(b1.w * scale);
  return f;
}

// ---------------- Prefill: causal flash attention, 1 wave per 16-query tile ----
// S^T = K x Q^T via WMMA (A<-K rows, B<-Q rows: identical per-lane packing),
// softmax per-lane (lane = query), P already lands in A-layout for O += P*V.
__global__ void __launch_bounds__(WAVES_PB * 32)
prefill_kernel(const float* __restrict__ q, const float* __restrict__ k,
               const float* __restrict__ v, float* __restrict__ out) {
#if USE_ASYNC
  __shared__ float sV[WAVES_PB][2][32 * VROW];   // double-buffered V tile (fp32)
#endif
  const int lane = threadIdx.x & 31;
  const int wib  = threadIdx.x >> 5;
  const int w    = blockIdx.x * WAVES_PB + wib;
  const int tilesPerHead = PROMPT_LEN / 16;                 // 128
  const int tile = w % tilesPerHead;
  const int h    = (w / tilesPerHead) % NUM_HEADS;
  const int b    = w / (tilesPerHead * NUM_HEADS);
  const int q0   = tile * 16;
  const int hi   = lane >> 4;
  const int lr   = lane & 15;

  // Q fragments (B-operand role for S^T = K * Q^T), pre-scaled by 1/sqrt(d)
  const int tokQ = b * PROMPT_LEN + q0 + lr;
  v16h aq[4];
#pragma unroll
  for (int c = 0; c < 4; ++c)
    aq[c] = load_frag(q + (size_t)tokQ * HD + h * HEAD_SIZE + 32 * c, hi, SCALE_F);

  v8f o[8];
#pragma unroll
  for (int t = 0; t < 8; ++t) o[t] = zero8();
  float m = NEG_BIG, l = 0.0f;

  const int kEnd = q0 + 16;                                  // causal key count
  int buf = 0;

#if USE_ASYNC
  // prologue: async-stage V tile for jb=0 into buffer 0 (one b128 per key row)
  {
    const float* src = v + (size_t)(b * PROMPT_LEN) * HD + h * HEAD_SIZE + lane * 4;
#pragma unroll
    for (int key = 0; key < 32; ++key)
      __builtin_amdgcn_global_load_async_to_lds_b128(
          (gvi4*)(src + (size_t)key * HD),
          (lvi4*)&sV[wib][0][key * VROW + lane * 4], 0, 0);
  }
#endif

  for (int jb = 0; jb < kEnd; jb += 32) {
    const int nj = jb + 32;
    const bool do_next = nj < kEnd;
    const int tokK = b * PROMPT_LEN + jb + lr;

#if USE_ASYNC
    if (do_next) {  // stage next V tile into the other buffer while we compute
      const float* src = v + (size_t)(b * PROMPT_LEN + nj) * HD + h * HEAD_SIZE + lane * 4;
#pragma unroll
      for (int key = 0; key < 32; ++key)
        __builtin_amdgcn_global_load_async_to_lds_b128(
            (gvi4*)(src + (size_t)key * HD),
            (lvi4*)&sV[wib][buf ^ 1][key * VROW + lane * 4], 0, 0);
    }
#endif
    if (do_next) {  // prefetch next K slab rows (global_prefetch_b8)
      __builtin_prefetch(k + (size_t)(tokK + 32) * HD + h * HEAD_SIZE, 0, 1);
      __builtin_prefetch(k + (size_t)(tokK + 48) * HD + h * HEAD_SIZE, 0, 1);
    }

    v8f s0 = zero8(), s1 = zero8();
#pragma unroll
    for (int c = 0; c < 4; ++c) {
      v16h klo = load_frag(k + (size_t)tokK * HD + h * HEAD_SIZE + 32 * c, hi, 1.0f);
      s0 = __builtin_amdgcn_wmma_f32_16x16x32_f16(false, klo, false, aq[c],
                                                  (short)0, s0, false, false);
    }
#pragma unroll
    for (int c = 0; c < 4; ++c) {
      v16h khi = load_frag(k + (size_t)(tokK + 16) * HD + h * HEAD_SIZE + 32 * c, hi, 1.0f);
      s1 = __builtin_amdgcn_wmma_f32_16x16x32_f16(false, khi, false, aq[c],
                                                  (short)0, s1, false, false);
    }
    // causal mask + local max (this lane owns query q0+lr; element r is key jb+r+8*hi / +16)
    float loc = NEG_BIG;
#pragma unroll
    for (int r = 0; r < 8; ++r) {
      int key0 = jb + r + 8 * hi;
      float x0 = (key0 <= q0 + lr) ? s0[r] : NEG_BIG;
      float x1 = (key0 + 16 <= q0 + lr) ? s1[r] : NEG_BIG;
      s0[r] = x0; s1[r] = x1;
      loc = fmaxf(loc, fmaxf(x0, x1));
    }
    float bm = fmaxf(loc, __shfl_xor(loc, 16, 32));
    float mn = fmaxf(m, bm);
    float corr = __expf(m - mn);
    float lsum = 0.0f;
    v16h pa;                                                  // P in A-layout, no shuffle needed
#pragma unroll
    for (int r = 0; r < 8; ++r) {
      float p0 = __expf(s0[r] - mn);
      float p1 = __expf(s1[r] - mn);
      lsum += p0 + p1;
      pa[r]     = (_Float16)p0;
      pa[8 + r] = (_Float16)p1;
    }
    lsum += __shfl_xor(lsum, 16, 32);
    l = l * corr + lsum;
    m = mn;
    // rescale accumulator: row M = r + 8*hi, its corr lives at lane (r + 8*hi)
#pragma unroll
    for (int r = 0; r < 8; ++r) {
      float cr = __shfl(corr, r + 8 * hi, 32);
#pragma unroll
      for (int t = 0; t < 8; ++t) o[t][r] *= cr;
    }

#if USE_ASYNC
    // current buffer's 32 async loads are the oldest outstanding ones
    if (do_next) __builtin_amdgcn_s_wait_asynccnt(32);
    else         __builtin_amdgcn_s_wait_asynccnt(0);
#endif

    // O += P * V ; V B-fragment: lane = output dim (t*16+lr), halves = keys
#pragma unroll
    for (int t = 0; t < 8; ++t) {
      v16h vb;
#if USE_ASYNC
#pragma unroll
      for (int e = 0; e < 8; ++e) {
        vb[e]     = (_Float16)sV[wib][buf][(e + 8 * hi) * VROW + t * 16 + lr];
        vb[8 + e] = (_Float16)sV[wib][buf][(e + 16 + 8 * hi) * VROW + t * 16 + lr];
      }
#else
      const float* vb_base = v + (size_t)(b * PROMPT_LEN + jb) * HD + h * HEAD_SIZE + t * 16 + lr;
#pragma unroll
      for (int e = 0; e < 8; ++e) {
        vb[e]     = (_Float16)vb_base[(size_t)(e + 8 * hi) * HD];
        vb[8 + e] = (_Float16)vb_base[(size_t)(e + 16 + 8 * hi) * HD];
      }
#endif
      o[t] = __builtin_amdgcn_wmma_f32_16x16x32_f16(false, pa, false, vb,
                                                    (short)0, o[t], false, false);
    }
    buf ^= 1;
  }
  // epilogue: divide by l (per query row M = r + 8*hi) and store
#pragma unroll
  for (int r = 0; r < 8; ++r) {
    float lrow = __shfl(l, r + 8 * hi, 32);
    float inv_l = 1.0f / lrow;
    int row = b * PROMPT_LEN + q0 + r + 8 * hi;
#pragma unroll
    for (int t = 0; t < 8; ++t)
      out[(size_t)row * HD + h * HEAD_SIZE + t * 16 + lr] = o[t][r] * inv_l;
  }
}

// ---------------- Inverse slot map (virtual reshape_and_cache) -----------------
__global__ void init_inv_kernel(int* __restrict__ inv, int n) {
  int i = blockIdx.x * blockDim.x + threadIdx.x;
  if (i < n) inv[i] = -1;
}
__global__ void fill_inv_kernel(const int* __restrict__ slot_mapping, int* __restrict__ inv, int n) {
  int t = blockIdx.x * blockDim.x + threadIdx.x;
  if (t < n) inv[slot_mapping[t]] = t;
}

// ---------------- Paged decode: 1 wave per (seq, head) -------------------------
__global__ void __launch_bounds__(256)
decode_kernel(const float* __restrict__ q, const float* __restrict__ knew,
              const float* __restrict__ vnew, const float* __restrict__ kcache,
              const float* __restrict__ vcache, const int* __restrict__ block_tables,
              const int* __restrict__ ctx_lens, const int* __restrict__ inv,
              float* __restrict__ out) {
  __shared__ float sQ[8][HEAD_SIZE];
  __shared__ float sS[8][MAX_BPS * BLOCK_SZ];
  const int lane = threadIdx.x & 31;
  const int wib  = threadIdx.x >> 5;
  const int w    = blockIdx.x * 8 + wib;
  const int n = w / NUM_HEADS, h = w % NUM_HEADS;
  const int qrow = P_TOKENS + n;
#pragma unroll
  for (int i = 0; i < 4; ++i)
    sQ[wib][lane + 32 * i] = q[(size_t)qrow * HD + h * HEAD_SIZE + lane + 32 * i];
  __syncthreads();

  const int ctx = ctx_lens[n];
  float m = NEG_BIG;
  for (int base = 0; base < ctx; base += 32) {
    int lkey = base + lane;
    float sc = NEG_BIG;
    if (lkey < ctx) {
      int blkid = block_tables[n * MAX_BPS + (lkey >> 4)];
      int off   = lkey & 15;
      int tok   = inv[blkid * BLOCK_SZ + off];
      float acc = 0.0f;
      if (tok >= 0) {
        const float* kp = knew + (size_t)tok * HD + h * HEAD_SIZE;
        for (int d = 0; d < HEAD_SIZE; ++d) acc += sQ[wib][d] * kp[d];
      } else {
        const float* kp = kcache + (((size_t)blkid * NUM_HEADS + h) * HEAD_SIZE) * BLOCK_SZ + off;
        for (int d = 0; d < HEAD_SIZE; ++d) acc += sQ[wib][d] * kp[(size_t)d * BLOCK_SZ];
      }
      sc = acc * SCALE_F;
      sS[wib][lkey] = sc;
    }
    m = fmaxf(m, sc);
  }
#pragma unroll
  for (int off = 16; off >= 1; off >>= 1) m = fmaxf(m, __shfl_xor(m, off, 32));
  float sum = 0.0f;
  for (int base = 0; base < ctx; base += 32) {
    int lkey = base + lane;
    if (lkey < ctx) {
      float p = __expf(sS[wib][lkey] - m);
      sS[wib][lkey] = p;
      sum += p;
    }
  }
#pragma unroll
  for (int off = 16; off >= 1; off >>= 1) sum += __shfl_xor(sum, off, 32);
  float inv_sum = 1.0f / sum;
  for (int dbase = 0; dbase < HEAD_SIZE; dbase += 32) {
    int d = dbase + lane;
    float acc = 0.0f;
    for (int lkey = 0; lkey < ctx; ++lkey) {
      float p = sS[wib][lkey];
      int blkid = block_tables[n * MAX_BPS + (lkey >> 4)];
      int off   = lkey & 15;
      int tok   = inv[blkid * BLOCK_SZ + off];
      float vv = (tok >= 0)
                     ? vnew[(size_t)tok * HD + h * HEAD_SIZE + d]
                     : vcache[(((size_t)blkid * NUM_HEADS + h) * HEAD_SIZE + d) * BLOCK_SZ + off];
      acc += p * vv;
    }
    out[(size_t)qrow * HD + h * HEAD_SIZE + d] = acc * inv_sum;
  }
}

extern "C" void kernel_launch(void* const* d_in, const int* in_sizes, int n_in,
                              void* d_out, int out_size, void* d_ws, size_t ws_size,
                              hipStream_t stream) {
  const float* query       = (const float*)d_in[0];
  const float* key         = (const float*)d_in[1];
  const float* value       = (const float*)d_in[2];
  const float* key_cache   = (const float*)d_in[3];
  const float* value_cache = (const float*)d_in[4];
  const int*   block_tbls  = (const int*)d_in[5];
  const int*   context_len = (const int*)d_in[6];
  const int*   slot_map    = (const int*)d_in[7];
  float* out = (float*)d_out;
  int*   inv = (int*)d_ws;   // NUM_BLOCKS*BLOCK_SZ ints

  const int nslots = NUM_BLOCKS * BLOCK_SZ;
  init_inv_kernel<<<(nslots + 255) / 256, 256, 0, stream>>>(inv, nslots);
  fill_inv_kernel<<<(T_TOKENS + 255) / 256, 256, 0, stream>>>(slot_map, inv, T_TOKENS);

  // 4096 tiles (2 prompts * 16 heads * 128 q-tiles), WAVES_PB waves per block
  prefill_kernel<<<4096 / WAVES_PB, WAVES_PB * 32, 0, stream>>>(query, key, value, out);

  // 512 (seq, head) waves, 8 waves per block
  decode_kernel<<<64, 256, 0, stream>>>(query, key, value, key_cache, value_cache,
                                        block_tbls, context_len, inv, out);
}
